// TensorDense_62113817035337
// MI455X (gfx1250) — matrix operand order repaired and checked by reading, not compiled, not run on hardware
//
#include <hip/hip_runtime.h>

typedef __attribute__((ext_vector_type(16))) _Float16 v16h;
typedef __attribute__((ext_vector_type(8)))  float    v8f;
typedef __attribute__((ext_vector_type(4)))  float    v4f;

#define S0    256
#define S1    256
#define BOND  32
#define BATCH 2048
#define UNITS (S0 * S1)

// ---------------------------------------------------------------------------
// Kernel 1: M_j = (core1[0]+core1[1]) @ core2[j]^T   (j = 0,1)
// Two 256x256x32 GEMMs. One wave computes one 16x16 tile with a single
// v_wmma_f32_16x16x32_f16 (BOND == 32 == WMMA K). 512 waves total.
// ---------------------------------------------------------------------------
__global__ __launch_bounds__(256) void precompute_M(const float* __restrict__ core1,
                                                    const float* __restrict__ core2,
                                                    float* __restrict__ M /* 2*256*256 */) {
  const int lane = threadIdx.x & 31;
  const int wid  = blockIdx.x * (blockDim.x >> 5) + (threadIdx.x >> 5); // 0..511
  const int j    = wid >> 8;        // which M matrix (0/1)
  const int tile = wid & 255;
  const int ti   = tile >> 4;       // row tile (a)
  const int tj   = tile & 15;       // col tile (c)
  const int g    = lane >> 4;       // lane half
  const int l16  = lane & 15;

  // A fragment (16x32, f16): A[a,p] = core1[0,a,p] + core1[1,a,p]
  // ISA layout: lane = row a; VGPR v<4 -> K = 8g+2v+{0,1}; v>=4 -> K = 16+8g+2(v-4)+{0,1}
  v16h a;
  const int arow = ti * 16 + l16;
  const float* c1a = core1 + arow * BOND;
  const float* c1b = core1 + S0 * BOND + arow * BOND;
#pragma unroll
  for (int e = 0; e < 16; ++e) {
    const int v = e >> 1, h = e & 1;
    const int k = ((v < 4) ? (8 * g + 2 * v) : (16 + 8 * g + 2 * (v - 4))) + h;
    a[e] = (_Float16)(c1a[k] + c1b[k]);
  }

  // B fragment (32x16, f16): B[p,c] = core2[j,c,p]
  // ISA layout: lane = column c; VGPR v<4 -> K = 8g+2v+{0,1}; v>=4 -> K = 16+8g+2(v-4)+{0,1}
  v16h b;
  const int ccol = tj * 16 + l16;
  const float* c2 = core2 + j * (S1 * BOND) + ccol * BOND;
#pragma unroll
  for (int e = 0; e < 16; ++e) {
    const int v = e >> 1, h = e & 1;
    b[e] = (_Float16)c2[((v < 4) ? (8 * g + 2 * v) : (16 + 8 * g + 2 * (v - 4))) + h];
  }

  v8f acc = {};
  acc = __builtin_amdgcn_wmma_f32_16x16x32_f16(
      /*neg_a=*/false, a, /*neg_b=*/false, b,
      /*c_mod=*/(short)0, acc, /*reuse_a=*/false, /*reuse_b=*/false);

  // C/D layout: VGPR v -> row v + 8g; lane -> column l16
  float* Mj = M + j * UNITS;
#pragma unroll
  for (int v = 0; v < 8; ++v) {
    Mj[(ti * 16 + v + 8 * g) * S1 + (tj * 16 + l16)] = acc[v];
  }
}

// ---------------------------------------------------------------------------
// Kernel 2: out[b] = relu(x0*M0 + x1*M1 + bias), streamed with NT stores.
// Each block owns a 1024-float slice of M0/M1/bias in registers and loops
// 32 batch indices -> 512 MB of pure non-temporal store traffic (HBM floor).
// ---------------------------------------------------------------------------
__global__ __launch_bounds__(256) void epilogue(const float* __restrict__ inp,
                                                const float* __restrict__ M,
                                                const float* __restrict__ bias,
                                                float* __restrict__ out) {
  const int off = blockIdx.x * 1024 + threadIdx.x * 4;   // slice within 65536
  const v4f m0 = *reinterpret_cast<const v4f*>(M + off);
  const v4f m1 = *reinterpret_cast<const v4f*>(M + UNITS + off);
  const v4f bz = *reinterpret_cast<const v4f*>(bias + off);

  for (int bb = blockIdx.y; bb < BATCH; bb += gridDim.y) {
    const float x0 = inp[2 * bb + 0];   // uniform -> scalar loads
    const float x1 = inp[2 * bb + 1];
    v4f r;
#pragma unroll
    for (int e = 0; e < 4; ++e)
      r[e] = fmaxf(fmaf(m0[e], x0, fmaf(m1[e], x1, bz[e])), 0.0f);
    __builtin_nontemporal_store(
        r, reinterpret_cast<v4f*>(out + (size_t)bb * UNITS + off));
  }
}

extern "C" void kernel_launch(void* const* d_in, const int* in_sizes, int n_in,
                              void* d_out, int out_size, void* d_ws, size_t ws_size,
                              hipStream_t stream) {
  (void)in_sizes; (void)n_in; (void)out_size; (void)ws_size;
  const float* inputs = (const float*)d_in[0];
  const float* core1  = (const float*)d_in[1];
  const float* core2  = (const float*)d_in[2];
  const float* bias   = (const float*)d_in[3];
  float* out = (float*)d_out;
  float* M   = (float*)d_ws;   // 2 * 256 * 256 floats = 512 KB scratch

  // 512 waves: one 16x16 WMMA tile each (2 matrices x 16x16 tiles)
  precompute_M<<<dim3(64), dim3(256), 0, stream>>>(core1, core2, M);
  // 64 slices x 64 batch-groups; each block reuses its M slice 32x
  epilogue<<<dim3(64, 64), dim3(256), 0, stream>>>(inputs, M, bias, out);
}
